// ModulatedGraphConv_59459527246233
// MI455X (gfx1250) — matrix operand order repaired and checked by reading, not compiled, not run on hardware
//
#include <hip/hip_runtime.h>
#include <hip/hip_bf16.h>

// ---------------------------------------------------------------------------
// ModulatedGraphConv fused kernel for MI455X (gfx1250, wave32, WMMA).
//
// out[b] = M ⊙ ( (diag⊙x[b]) @ W0 + (A_off @ x[b]) @ W1 ) + bias
//
// Single fused GEMM with logical K = 512 (both W-halves share one staged x
// tile). bf16 hi/lo split (3 WMMA terms) gives ~fp32 accuracy at bf16 rate.
// x is read from HBM once (285 MB), out written once (285 MB).
// z (GEMM A operand) is converted to packed bf16 fragment layout ONCE in the
// z-compute phase, so the WMMA loop does only ds_load_b128 + v_wmma.
// ---------------------------------------------------------------------------

typedef __attribute__((ext_vector_type(16))) __bf16 v16bf;
typedef __attribute__((ext_vector_type(8)))  float  v8f;

#define B_TOT    16384
#define J_N      17
#define F_N      256
#define ROWS     (B_TOT * J_N)     // 278528 = 2176 * 128 (exact)
#define WG_ROWS  128
#define KTILES   8                 // 8 x-K-tiles of 32 (each used for W0 & W1)
#define XSTRIDE  36                // padded f32 row stride (16B aligned)
#define ZSTR_DW  20                // z row stride in dwords (16B aligned, no conflicts)
#define ST_ROWS  153               // 9 batches * 17 joints staged per tile

// workspace layout (32-bit units):
//   [0 .. 131072)  : packed W fragments: [kt][hi(8192)|lo(8192)] dwords,
//                    dword index r = mat*4096 + nt*256 + lane*8 + v
//   [131072 .. )   : diag[17] floats
//   [131104 .. )   : A_off[289] floats
#define WS_DIAG   131072
#define WS_AOFF   131104

static __device__ __forceinline__ unsigned short bf_bits(__bf16 h) {
    return __builtin_bit_cast(unsigned short, h);
}

static __device__ __forceinline__ unsigned pack_bf16_pair(float f0, float f1) {
    return (unsigned)bf_bits((__bf16)f0) | ((unsigned)bf_bits((__bf16)f1) << 16);
}

// ---------------------------------------------------------------------------
// Setup kernel 1: convert W[2][256][256] f32 -> hi/lo bf16 fragment-linear
// layout matching the 16-bit B-matrix (32x16) VGPR layout:
//   lane l : N = 16*nt + (l&15) ; dword v : K = 16*(v>>2) + 8*(l>>4) + 2*(v&3),+1
// ---------------------------------------------------------------------------
__global__ __launch_bounds__(256) void mgc_pack_w(const float* __restrict__ W,
                                                  unsigned* __restrict__ wpk) {
    int idx = blockIdx.x * 256 + threadIdx.x;        // 0 .. 65535
    int kt   = idx >> 13;                            // 0..7
    int r    = idx & 8191;                           // dword index within kt
    int mat  = r >> 12;                              // 0..1
    int r2   = r & 4095;
    int nt   = r2 >> 8;                              // 0..15
    int r3   = r2 & 255;
    int lane = r3 >> 3;                              // 0..31
    int v    = r3 & 7;                               // 0..7
    int k0   = kt * 32 + ((v >> 2) << 4) + ((lane >> 4) << 3) + 2 * (v & 3); // K of low half
    int n    = nt * 16 + (lane & 15);

    float w0 = W[(mat * F_N + k0)     * F_N + n];
    float w1 = W[(mat * F_N + k0 + 1) * F_N + n];
    __bf16 h0 = (__bf16)w0, h1 = (__bf16)w1;
    float  r0 = w0 - (float)h0, r1 = w1 - (float)h1;

    wpk[kt * 16384 + r]        = (unsigned)bf_bits(h0) | ((unsigned)bf_bits(h1) << 16);
    wpk[kt * 16384 + 8192 + r] = pack_bf16_pair(r0, r1);
}

// ---------------------------------------------------------------------------
// Setup kernel 2: symmetrized adjacency -> diag[17] + zero-diagonal A_off[289]
// ---------------------------------------------------------------------------
__global__ __launch_bounds__(256) void mgc_prep_adj(const float* __restrict__ adj,
                                                    const float* __restrict__ adj2,
                                                    float* __restrict__ wsf) {
    for (int idx = threadIdx.x; idx < J_N * J_N; idx += 256) {
        int i = idx / J_N, j = idx % J_N;
        float aij = adj[i * J_N + j] + adj2[i * J_N + j];
        float aji = adj[j * J_N + i] + adj2[j * J_N + i];
        float v = 0.5f * (aij + aji);
        if (i == j) { wsf[WS_DIAG + i] = v; v = 0.0f; }
        wsf[WS_AOFF + idx] = v;
    }
}

// ---------------------------------------------------------------------------
// Main fused kernel. 512 threads = 16 waves; wave grid 4(M) x 4(N).
// Workgroup tile: 128 rows x 256 cols. Per wave: 2 mtiles x 4 ntiles,
// acc = 8 x v8f = 64 VGPRs.
// ---------------------------------------------------------------------------
__global__ __launch_bounds__(512) void mgc_main(const float* __restrict__ x,
                                                const float* __restrict__ Mmat,
                                                const float* __restrict__ bias,
                                                const unsigned* __restrict__ ws_u,
                                                const float* __restrict__ ws_f,
                                                float* __restrict__ out) {
    extern __shared__ char smem[];
    float*    xst   = (float*)smem;                          // 153*36 f32
    // z planes: [zi(2)][hilo(2)][128 rows][20 dw], packed bf16 K-pairs
    unsigned* zz    = (unsigned*)(xst + ST_ROWS * XSTRIDE);
    unsigned* bst   = zz + 4 * WG_ROWS * ZSTR_DW;            // 16384 dw W stage
    float*    adiag = (float*)(bst + 16384);                 // 32 f32 (17 used)
    float*    aoff  = adiag + 32;                            // 304 f32 (289 used)

    const int tid  = threadIdx.x;
    const int lane = tid & 31;
    const int wave = tid >> 5;
    const int l16  = lane & 15;
    const int lhi  = lane >> 4;           // 0/1 half-wave
    const int wm   = wave & 3;            // M group: rows wm*32 .. +31
    const int wn   = wave >> 2;           // N group: cols wn*64 .. +63

    const int r0     = blockIdx.x * WG_ROWS;
    const int bs0    = r0 / J_N;          // first batch touched by this tile
    const int stRow0 = bs0 * J_N;         // first staged global (b,j) row

    // preload adjacency into LDS (visible after the first in-loop barrier)
    for (int idx = tid; idx < J_N; idx += 512)        adiag[idx] = ws_f[WS_DIAG + idx];
    for (int idx = tid; idx < J_N * J_N; idx += 512)  aoff[idx]  = ws_f[WS_AOFF + idx];

    v8f acc[2][4];
#pragma unroll
    for (int mt = 0; mt < 2; ++mt)
#pragma unroll
        for (int nt = 0; nt < 4; ++nt)
#pragma unroll
            for (int e = 0; e < 8; ++e) acc[mt][nt][e] = 0.0f;

    for (int kt = 0; kt < KTILES; ++kt) {
        const int kx = kt * 32;
        __syncthreads();   // previous iteration's LDS consumers done

        // ---- stage x rows [stRow0, stRow0+153) for this K-slice (b128) ----
        for (int idx = tid; idx < ST_ROWS * 8; idx += 512) {
            int row = idx >> 3, c = idx & 7;
            int gr = stRow0 + row;
            float4 val = make_float4(0.f, 0.f, 0.f, 0.f);
            if (gr < ROWS)
                val = *(const float4*)(x + (size_t)gr * F_N + kx + c * 4);
            *(float4*)(xst + row * XSTRIDE + c * 4) = val;
        }
        // ---- stage packed W fragments (hi+lo, both mats) for this K-slice ----
        {
            const uint4* src = (const uint4*)(ws_u + kt * 16384);
            uint4*       dst = (uint4*)bst;
            for (int idx = tid; idx < 4096; idx += 512) dst[idx] = src[idx];
        }
        __syncthreads();

        // ---- build z0 = diag⊙x, z1 = A_off-mixed rows; convert to packed
        //      bf16 hi/lo A-fragment layout ONCE (dword c = K pair 2c,2c+1) ----
        {
            const int c    = tid & 15;        // K-pair index 0..15
            const int rowg = tid >> 4;        // 0..31
#pragma unroll
            for (int rep = 0; rep < 4; ++rep) {
                int row  = rowg + rep * 32;    // 0..127
                int gr   = r0 + row;
                int j    = gr % J_N;
                int srow = gr - stRow0;        // staged row of (b, j)
                int b17  = srow - j;           // staged row of (b, 0)

                float2 xv = *(const float2*)(xst + srow * XSTRIDE + 2 * c);
                float d   = adiag[j];
                float z0a = d * xv.x, z0b = d * xv.y;

                float z1a = 0.f, z1b = 0.f;
#pragma unroll
                for (int j2 = 0; j2 < J_N; ++j2) {
                    float2 xj = *(const float2*)(xst + (b17 + j2) * XSTRIDE + 2 * c);
                    float  a  = aoff[j * J_N + j2];
                    z1a += a * xj.x;
                    z1b += a * xj.y;
                }

                // hi planes
                __bf16 h0a = (__bf16)z0a, h0b = (__bf16)z0b;
                __bf16 h1a = (__bf16)z1a, h1b = (__bf16)z1b;
                unsigned* zr0 = zz + (0 * WG_ROWS + row) * ZSTR_DW;            // z0 hi
                unsigned* zr1 = zz + (1 * WG_ROWS + row) * ZSTR_DW;            // z0 lo
                unsigned* zr2 = zz + (2 * WG_ROWS + row) * ZSTR_DW;            // z1 hi
                unsigned* zr3 = zz + (3 * WG_ROWS + row) * ZSTR_DW;            // z1 lo
                zr0[c] = (unsigned)bf_bits(h0a) | ((unsigned)bf_bits(h0b) << 16);
                zr2[c] = (unsigned)bf_bits(h1a) | ((unsigned)bf_bits(h1b) << 16);
                zr1[c] = pack_bf16_pair(z0a - (float)h0a, z0b - (float)h0b);
                zr3[c] = pack_bf16_pair(z1a - (float)h1a, z1b - (float)h1b);
            }
        }
        __syncthreads();

        // ---- WMMA: z0@W0 + z1@W1, hi/lo 3-term split (pure ds_load + wmma) ----
#pragma unroll
        for (int mat = 0; mat < 2; ++mat) {
            // A fragments: 16-bit A 16x32 layout -> element chunk [0..7] is
            // K = 8*lhi..+7 (dwords 4*lhi..+3), chunk [8..15] is K = 16+8*lhi..+7
            // (dwords 8+4*lhi..+3).
            v16bf ah[2], al[2];
#pragma unroll
            for (int mt = 0; mt < 2; ++mt) {
                int row = wm * 32 + mt * 16 + l16;
                const unsigned* ph = zz + ((2 * mat + 0) * WG_ROWS + row) * ZSTR_DW;
                const unsigned* pl = zz + ((2 * mat + 1) * WG_ROWS + row) * ZSTR_DW;
                union { uint4 q[2]; v16bf v; } th, tl;
                th.q[0] = *(const uint4*)(ph + 4 * lhi);
                th.q[1] = *(const uint4*)(ph + 8 + 4 * lhi);
                tl.q[0] = *(const uint4*)(pl + 4 * lhi);
                tl.q[1] = *(const uint4*)(pl + 8 + 4 * lhi);
                ah[mt] = th.v;
                al[mt] = tl.v;
            }
#pragma unroll
            for (int nt = 0; nt < 4; ++nt) {
                union { uint4 q[2]; v16bf v; } bh, bl;
                int fi = ((mat * 16 + wn * 4 + nt) * 32 + lane) * 8;
                const uint4* bp = (const uint4*)(bst + fi);
                const uint4* bq = (const uint4*)(bst + 8192 + fi);
                bh.q[0] = bp[0]; bh.q[1] = bp[1];
                bl.q[0] = bq[0]; bl.q[1] = bq[1];
#pragma unroll
                for (int mt = 0; mt < 2; ++mt) {
                    acc[mt][nt] = __builtin_amdgcn_wmma_f32_16x16x32_bf16(
                        false, ah[mt], false, bh.v, (short)0, acc[mt][nt], false, false);
                    acc[mt][nt] = __builtin_amdgcn_wmma_f32_16x16x32_bf16(
                        false, al[mt], false, bh.v, (short)0, acc[mt][nt], false, false);
                    acc[mt][nt] = __builtin_amdgcn_wmma_f32_16x16x32_bf16(
                        false, ah[mt], false, bl.v, (short)0, acc[mt][nt], false, false);
                }
            }
        }
    }

    // ---- epilogue: out = acc * M[j, col] + bias[col]  (C layout: row=v+8*lhi) ----
#pragma unroll
    for (int mt = 0; mt < 2; ++mt) {
        int rowbase = r0 + wm * 32 + mt * 16 + lhi * 8;
#pragma unroll
        for (int nt = 0; nt < 4; ++nt) {
            int col = wn * 64 + nt * 16 + l16;
            float bv = bias[col];
#pragma unroll
            for (int v = 0; v < 8; ++v) {
                int gr = rowbase + v;
                int j  = gr % J_N;
                out[(size_t)gr * F_N + col] =
                    acc[mt][nt][v] * Mmat[j * F_N + col] + bv;
            }
        }
    }
}

// ---------------------------------------------------------------------------
// Host launcher
// ---------------------------------------------------------------------------
extern "C" void kernel_launch(void* const* d_in, const int* in_sizes, int n_in,
                              void* d_out, int out_size, void* d_ws, size_t ws_size,
                              hipStream_t stream) {
    const float* x    = (const float*)d_in[0];   // [16384,17,256]
    const float* W    = (const float*)d_in[1];   // [2,256,256]
    const float* M    = (const float*)d_in[2];   // [17,256]
    const float* adj  = (const float*)d_in[3];   // [17,17]
    const float* adj2 = (const float*)d_in[4];   // [17,17]
    const float* bias = (const float*)d_in[5];   // [256]
    float*       out  = (float*)d_out;

    unsigned* ws_u = (unsigned*)d_ws;
    float*    ws_f = (float*)d_ws;

    // 1) pack W into hi/lo bf16 fragment layout (131072 dwords)
    mgc_pack_w<<<256, 256, 0, stream>>>(W, ws_u);
    // 2) symmetrize adjacency -> diag + off-diagonal
    mgc_prep_adj<<<1, 256, 0, stream>>>(adj, adj2, ws_f);

    // 3) fused GEMM + joint-mix + epilogue
    constexpr int smem_bytes =
        (ST_ROWS * XSTRIDE) * 4 +            // x stage
        (4 * WG_ROWS * ZSTR_DW) * 4 +        // z bf16 hi/lo planes
        16384 * 4 +                          // W fragment stage
        (32 + 304) * 4;                      // diag + A_off
    mgc_main<<<ROWS / WG_ROWS, 512, smem_bytes, stream>>>(x, M, bias, ws_u, ws_f, out);
}
